// FineTuneEsmCNN_49967649521953
// MI455X (gfx1250) — matrix operand order repaired and checked by reading, not compiled, not run on hardware
//
#include <hip/hip_runtime.h>
#include <hip/hip_bf16.h>

typedef __attribute__((ext_vector_type(16))) __bf16 v16bf;
typedef __attribute__((ext_vector_type(8)))  __bf16 v8bf;
typedef __attribute__((ext_vector_type(8)))  float  v8f;

namespace {
constexpr int NB = 8, NT = 4096, DIN = 1330, DH = 64, DC = 16;
constexpr int MTOT = NB * NT;      // 32768 tokens
constexpr int NSTEP = NT - 1;      // 4095 scan steps (t = 1..T-1)
constexpr int CCH = 64, CLEN = 64; // chunked associative scan
constexpr int KCH = 42;            // ceil(1330/32) K-chunks for GEMM1
constexpr float FNEG = -1e30f;

// workspace byte offsets (all 32B-aligned)
constexpr size_t OFF_H16  = 0;                          // 32768*64*2   = 4,194,304
constexpr size_t OFF_EM   = OFF_H16 + 4194304;          // 32768*2*4    =   262,144
constexpr size_t OFF_MATS = OFF_EM + 262144;            // 8*64*8*4     =    16,384
constexpr size_t OFF_CST  = OFF_MATS + 16384;           // 8*64*4*4     =     8,192
constexpr size_t OFF_WF   = OFF_CST + 8192;             // 42*2048*2    =   172,032
constexpr size_t OFF_CF   = OFF_WF + 172032;            // 6*512*2      =     6,144
constexpr size_t OFF_NUM  = OFF_CF + 6144;              // 32
constexpr size_t OFF_DEN  = OFF_NUM + 32;               // 32
constexpr size_t OFF_LT   = OFF_DEN + 32;               // 32
constexpr size_t OFF_HIST = OFF_LT + 32;                // 8*4095 bytes
}

__device__ __forceinline__ float lse2(float a, float b) {
  float m = fmaxf(a, b);
  return m + __logf(__expf(a - m) + __expf(b - m));
}

// ---------------------------------------------------------------------------
// Prep: W1 -> bf16, B-fragment order, zero-padded to 42 chunks.
// wf[c][idx] : idx -> elem=idx&15, lane=(idx>>4)&31, tile=idx>>9
// ---------------------------------------------------------------------------
__global__ void k_prep_w1(const float* __restrict__ W1, __bf16* __restrict__ wf) {
  int idx = blockIdx.x * 256 + threadIdx.x;
  if (idx >= KCH * 2048) return;
  int c = idx >> 11, r = idx & 2047;
  int elem = r & 15, ln = (r >> 4) & 31, tt = r >> 9;
  int k = c * 32 + (ln >> 4) * 8 + (elem & 7) + (elem >> 3) * 16;
  int n = tt * 16 + (ln & 15);
  wf[idx] = (k < DIN) ? (__bf16)W1[(size_t)n * DIN + k] : (__bf16)0.0f;
}

// Prep: Wc -> bf16 B-fragments for the conv GEMM (K=192 = 6 chunks, N=16).
__global__ void k_prep_wc(const float* __restrict__ Wc, __bf16* __restrict__ cf) {
  int idx = blockIdx.x * 256 + threadIdx.x;
  if (idx >= 6 * 512) return;
  int c = idx >> 9, r = idx & 511;
  int elem = r & 15, ln = r >> 4;
  int k = c * 32 + (ln >> 4) * 8 + (elem & 7) + (elem >> 3) * 16;
  int w = k >> 6, ch = k & 63;
  int n = ln & 15;
  cf[idx] = (__bf16)Wc[(size_t)n * 192 + ch * 3 + w];
}

// ---------------------------------------------------------------------------
// Kernel 1: h = relu(x @ W1^T + b1)  (M=32768, K=1330, N=64), bf16 WMMA.
// Double-buffered LDS staging of pre-fragmented W1; unpredicated float2 A
// loads (tail chunk peeled by uniform branch). h stored as bf16.
// ---------------------------------------------------------------------------
__global__ void k_gemm1(const float* __restrict__ x, const __bf16* __restrict__ wf,
                        const float* __restrict__ b1, __bf16* __restrict__ h16) {
  __shared__ v16bf lb[2][4 * 32]; // [buf][n-tile*32 + lane]
  const int tid = threadIdx.x;
  const int lane = tid & 31;
  const int wave = tid >> 5;
  const int kSel = lane >> 4;
  const int mBase = (blockIdx.x * 8 + wave) * 16;
  const int row = mBase + (lane & 15);
  const float*  xr  = x + (size_t)row * DIN;
  const float2* xr2 = (const float2*)xr;           // row*1330 even -> 8B aligned
  const uint4*  wf4 = (const uint4*)wf;

  v8f acc0 = {}, acc1 = {}, acc2 = {}, acc3 = {};

  ((uint4*)lb[0])[tid] = wf4[tid];                 // stage chunk 0

  for (int c = 0; c < KCH; ++c) {
    __syncthreads();                               // chunk c staged & prev reads done
    if (c + 1 < KCH)
      ((uint4*)lb[(c + 1) & 1])[tid] = wf4[(size_t)(c + 1) * 256 + tid];

    const int k0 = c * 32;
    const int ka = k0 + kSel * 8;
    const int kb = k0 + 16 + kSel * 8;
    v16bf a;
    if (c != KCH - 1) {
      float2 p0 = xr2[(ka >> 1) + 0], p1 = xr2[(ka >> 1) + 1];
      float2 p2 = xr2[(ka >> 1) + 2], p3 = xr2[(ka >> 1) + 3];
      float2 q0 = xr2[(kb >> 1) + 0], q1 = xr2[(kb >> 1) + 1];
      float2 q2 = xr2[(kb >> 1) + 2], q3 = xr2[(kb >> 1) + 3];
      a[0] = (__bf16)p0.x; a[1] = (__bf16)p0.y; a[2] = (__bf16)p1.x; a[3] = (__bf16)p1.y;
      a[4] = (__bf16)p2.x; a[5] = (__bf16)p2.y; a[6] = (__bf16)p3.x; a[7] = (__bf16)p3.y;
      a[8] = (__bf16)q0.x; a[9] = (__bf16)q0.y; a[10] = (__bf16)q1.x; a[11] = (__bf16)q1.y;
      a[12] = (__bf16)q2.x; a[13] = (__bf16)q2.y; a[14] = (__bf16)q3.x; a[15] = (__bf16)q3.y;
    } else {
      // tail chunk k0=1312: segment A (max k=1327) fully valid; segment B ragged
      float2 p0 = xr2[(ka >> 1) + 0], p1 = xr2[(ka >> 1) + 1];
      float2 p2 = xr2[(ka >> 1) + 2], p3 = xr2[(ka >> 1) + 3];
      a[0] = (__bf16)p0.x; a[1] = (__bf16)p0.y; a[2] = (__bf16)p1.x; a[3] = (__bf16)p1.y;
      a[4] = (__bf16)p2.x; a[5] = (__bf16)p2.y; a[6] = (__bf16)p3.x; a[7] = (__bf16)p3.y;
#pragma unroll
      for (int i = 0; i < 8; ++i) {
        int k2 = kb + i;
        a[8 + i] = (k2 < DIN) ? (__bf16)xr[k2] : (__bf16)0.0f;
      }
    }
    const v16bf* lbc = lb[c & 1];
    acc0 = __builtin_amdgcn_wmma_f32_16x16x32_bf16(false, a, false, lbc[0 * 32 + lane], (short)0, acc0, false, false);
    acc1 = __builtin_amdgcn_wmma_f32_16x16x32_bf16(false, a, false, lbc[1 * 32 + lane], (short)0, acc1, false, false);
    acc2 = __builtin_amdgcn_wmma_f32_16x16x32_bf16(false, a, false, lbc[2 * 32 + lane], (short)0, acc2, false, false);
    acc3 = __builtin_amdgcn_wmma_f32_16x16x32_bf16(false, a, false, lbc[3 * 32 + lane], (short)0, acc3, false, false);
  }

  // C/D layout: VGPR r -> M = r + 8*(lane>=16), N = lane&15 per tile
  const int hi = lane >> 4;
  const int nlo = lane & 15;
#pragma unroll
  for (int r = 0; r < 8; ++r) {
    int m = mBase + r + hi * 8;
    __bf16* hr = h16 + (size_t)m * DH + nlo;
    hr[0]  = (__bf16)fmaxf(acc0[r] + b1[nlo],      0.0f);
    hr[16] = (__bf16)fmaxf(acc1[r] + b1[16 + nlo], 0.0f);
    hr[32] = (__bf16)fmaxf(acc2[r] + b1[32 + nlo], 0.0f);
    hr[48] = (__bf16)fmaxf(acc3[r] + b1[48 + nlo], 0.0f);
  }
}

// ---------------------------------------------------------------------------
// Kernel 2: width-3 time conv as WMMA GEMM (K=192, N=16) + ReLU + tiny GEMM2
// ---------------------------------------------------------------------------
__global__ void k_conv_em(const __bf16* __restrict__ h16, const __bf16* __restrict__ cf,
                          const float* __restrict__ bc, const float* __restrict__ W2,
                          const float* __restrict__ b2, float* __restrict__ em) {
  __shared__ float hl[8][16][16];
  const int lane = threadIdx.x & 31;
  const int wave = threadIdx.x >> 5;
  const int kSel = lane >> 4;
  const int mBase = (blockIdx.x * 8 + wave) * 16;
  const int m = mBase + (lane & 15);
  const int t = m & (NT - 1);
  const __bf16* hb = h16 + (size_t)(m - t) * DH;  // batch base
  const v16bf* cfv = (const v16bf*)cf;

  v8f acc = {};
#pragma unroll
  for (int k0 = 0; k0 < 192; k0 += 32) {
    v8bf s0 = {}, s1 = {};
    {
      int kq = k0 + kSel * 8;
      int w = kq >> 6, ch = kq & 63;
      int tp = t - 1 + w;
      if (tp >= 0 && tp < NT) s0 = *(const v8bf*)(hb + (size_t)tp * DH + ch);
    }
    {
      int kq = k0 + 16 + kSel * 8;
      int w = kq >> 6, ch = kq & 63;
      int tp = t - 1 + w;
      if (tp >= 0 && tp < NT) s1 = *(const v8bf*)(hb + (size_t)tp * DH + ch);
    }
    v16bf a = __builtin_shufflevector(s0, s1, 0, 1, 2, 3, 4, 5, 6, 7,
                                      8, 9, 10, 11, 12, 13, 14, 15);
    v16bf bfrag = cfv[(k0 >> 5) * 32 + lane];
    acc = __builtin_amdgcn_wmma_f32_16x16x32_bf16(false, a, false, bfrag, (short)0, acc, false, false);
  }
  const int hi = lane >> 4, cN = lane & 15;
#pragma unroll
  for (int r = 0; r < 8; ++r)
    hl[wave][r + hi * 8][cN] = fmaxf(acc[r] + bc[cN], 0.0f);
  __syncthreads();
  if (lane < 16) {
    int mm = mBase + lane;
    float e0 = b2[0], e1 = b2[1];
#pragma unroll
    for (int c = 0; c < DC; ++c) {
      float v = hl[wave][lane][c];
      e0 += v * W2[c];
      e1 += v * W2[DC + c];
    }
    em[(size_t)mm * 2 + 0] = e0;
    em[(size_t)mm * 2 + 1] = e1;
  }
}

// ---------------------------------------------------------------------------
// Kernel 3: CRF numerator per batch (parallel reduction over t)
// ---------------------------------------------------------------------------
__global__ void k_crf_num(const float* __restrict__ em, const int* __restrict__ labels,
                          const int* __restrict__ tlen, const float* __restrict__ startv,
                          const float* __restrict__ endv, const float* __restrict__ trans,
                          float* __restrict__ num) {
  __shared__ float red[256];
  const int b = blockIdx.x;
  const int len = tlen[b];
  const int* tg = labels + (size_t)b * NT;
  const float* eb = em + (size_t)b * NT * 2;
  float part = 0.0f;
  for (int t = 1 + (int)threadIdx.x; t < NT; t += 256) {
    if (t < len) {
      int pt = tg[t - 1], ct = tg[t];
      part += eb[t * 2 + ct] + trans[pt * 2 + ct];
    }
  }
  red[threadIdx.x] = part;
  __syncthreads();
  for (int s = 128; s > 0; s >>= 1) {
    if ((int)threadIdx.x < s) red[threadIdx.x] += red[threadIdx.x + s];
    __syncthreads();
  }
  if (threadIdx.x == 0) {
    int t0 = tg[0];
    num[b] = red[0] + startv[t0] + eb[t0] + endv[tg[len - 1]];
  }
}

// ---------------------------------------------------------------------------
// Kernel 4: per-chunk 2x2 transfer matrices, log semiring (llh) and
// max-plus semiring (Viterbi). Masked steps = identity (skipped).
// ---------------------------------------------------------------------------
__global__ void k_scan_chunks(const float* __restrict__ em, const int* __restrict__ tlen,
                              const float* __restrict__ trans, float* __restrict__ mats) {
  int g = blockIdx.x * blockDim.x + threadIdx.x;
  if (g >= NB * CCH) return;
  int b = g / CCH, c = g % CCH;
  int len = tlen[b];
  const float* eb = em + (size_t)b * NT * 2;
  const float t00 = trans[0], t01 = trans[1], t10 = trans[2], t11 = trans[3];
  float S00 = 0, S01 = FNEG, S10 = FNEG, S11 = 0;
  float V00 = 0, V01 = FNEG, V10 = FNEG, V11 = 0;
  int s0 = c * CLEN;
  int s1 = s0 + CLEN; if (s1 > NSTEP) s1 = NSTEP;
  for (int s = s0; s < s1; ++s) {
    int t = s + 1;
    if (t < len) {
      float e0 = eb[t * 2], e1 = eb[t * 2 + 1];
      float m00 = t00 + e0, m01 = t01 + e1, m10 = t10 + e0, m11 = t11 + e1;
      float nS00 = lse2(S00 + m00, S01 + m10);
      float nS01 = lse2(S00 + m01, S01 + m11);
      float nS10 = lse2(S10 + m00, S11 + m10);
      float nS11 = lse2(S10 + m01, S11 + m11);
      S00 = nS00; S01 = nS01; S10 = nS10; S11 = nS11;
      float nV00 = fmaxf(V00 + m00, V01 + m10);
      float nV01 = fmaxf(V00 + m01, V01 + m11);
      float nV10 = fmaxf(V10 + m00, V11 + m10);
      float nV11 = fmaxf(V10 + m01, V11 + m11);
      V00 = nV00; V01 = nV01; V10 = nV10; V11 = nV11;
    }
  }
  float* o = mats + (size_t)g * 8;
  o[0] = S00; o[1] = S01; o[2] = S10; o[3] = S11;
  o[4] = V00; o[5] = V01; o[6] = V10; o[7] = V11;
}

// ---------------------------------------------------------------------------
// Kernel 5: sequential combine over 64 chunk matrices per batch.
// ---------------------------------------------------------------------------
__global__ void k_scan_combine(const float* __restrict__ em,
                               const float* __restrict__ startv, const float* __restrict__ endv,
                               const float* __restrict__ mats, float* __restrict__ chunk_start,
                               float* __restrict__ den, int* __restrict__ lastTag) {
  int b = threadIdx.x;
  if (b >= NB) return;
  const float* eb = em + (size_t)b * NT * 2;
  float s0v = startv[0] + eb[0], s1v = startv[1] + eb[1];
  float v0 = s0v, v1 = s1v;
  for (int c = 0; c < CCH; ++c) {
    float* cs = chunk_start + (size_t)(b * CCH + c) * 4;
    cs[0] = s0v; cs[1] = s1v; cs[2] = v0; cs[3] = v1;
    const float* Mx = mats + (size_t)(b * CCH + c) * 8;
    float ns0 = lse2(s0v + Mx[0], s1v + Mx[2]);
    float ns1 = lse2(s0v + Mx[1], s1v + Mx[3]);
    float nv0 = fmaxf(v0 + Mx[4], v1 + Mx[6]);
    float nv1 = fmaxf(v0 + Mx[5], v1 + Mx[7]);
    s0v = ns0; s1v = ns1; v0 = nv0; v1 = nv1;
  }
  den[b] = lse2(s0v + endv[0], s1v + endv[1]);
  lastTag[b] = (v1 + endv[1] > v0 + endv[0]) ? 1 : 0;
}

// ---------------------------------------------------------------------------
// Kernel 6: parallel Viterbi replay, packed 2-bit backpointers per step.
// ---------------------------------------------------------------------------
__global__ void k_vit_replay(const float* __restrict__ em, const int* __restrict__ tlen,
                             const float* __restrict__ trans, const float* __restrict__ chunk_start,
                             unsigned char* __restrict__ hist) {
  int g = blockIdx.x * blockDim.x + threadIdx.x;
  if (g >= NB * CCH) return;
  int b = g / CCH, c = g % CCH;
  int len = tlen[b];
  const float* eb = em + (size_t)b * NT * 2;
  const float t00 = trans[0], t01 = trans[1], t10 = trans[2], t11 = trans[3];
  const float* cs = chunk_start + (size_t)g * 4;
  float v0 = cs[2], v1 = cs[3];
  unsigned char* hb = hist + (size_t)b * NSTEP;
  int s0 = c * CLEN, s1 = s0 + CLEN; if (s1 > NSTEP) s1 = NSTEP;
  for (int s = s0; s < s1; ++s) {
    int t = s + 1;
    int bp0 = (v1 + t10 > v0 + t00) ? 1 : 0;
    int bp1 = (v1 + t11 > v0 + t01) ? 1 : 0;
    hb[s] = (unsigned char)(bp0 | (bp1 << 1));
    if (t < len) {
      float e0 = eb[t * 2], e1 = eb[t * 2 + 1];
      float nv0 = fmaxf(v0 + t00, v1 + t10) + e0;
      float nv1 = fmaxf(v0 + t01, v1 + t11) + e1;
      v0 = nv0; v1 = nv1;
    }
  }
}

// ---------------------------------------------------------------------------
// Kernel 7: backtrace per batch + final -llh
// ---------------------------------------------------------------------------
__global__ void k_backtrace(const unsigned char* __restrict__ hist, const int* __restrict__ tlen,
                            const int* __restrict__ lastTag, const float* __restrict__ num,
                            const float* __restrict__ den, float* __restrict__ out) {
  int b = threadIdx.x;
  if (b < NB) {
    int len = tlen[b];
    int tag = lastTag[b];
    float* ot = out + 1 + (size_t)b * NT;
    const unsigned char* hb = hist + (size_t)b * NSTEP;
    for (int t = NT - 1; t >= 1; --t) {
      ot[t] = (float)tag;
      int prev = (hb[t - 1] >> tag) & 1;
      if (t < len) tag = prev;
    }
    ot[0] = (float)tag;
  } else if (b == NB) {
    float acc = 0.0f;
    for (int i = 0; i < NB; ++i) acc += num[i] - den[i];
    out[0] = -acc;
  }
}

// ---------------------------------------------------------------------------
extern "C" void kernel_launch(void* const* d_in, const int* in_sizes, int n_in,
                              void* d_out, int out_size, void* d_ws, size_t ws_size,
                              hipStream_t stream) {
  (void)in_sizes; (void)n_in; (void)out_size; (void)ws_size;
  const float* x      = (const float*)d_in[0];
  const int*   tlen   = (const int*)d_in[1];
  const int*   labels = (const int*)d_in[2];
  const float* W1     = (const float*)d_in[3];
  const float* b1     = (const float*)d_in[4];
  const float* Wc     = (const float*)d_in[5];
  const float* bc     = (const float*)d_in[6];
  const float* W2     = (const float*)d_in[7];
  const float* b2     = (const float*)d_in[8];
  const float* startv = (const float*)d_in[9];
  const float* endv   = (const float*)d_in[10];
  const float* trans  = (const float*)d_in[11];
  float* out = (float*)d_out;

  char* base = (char*)d_ws;
  __bf16* h16 = (__bf16*)(base + OFF_H16);
  float*  em  = (float*)(base + OFF_EM);
  float*  mats = (float*)(base + OFF_MATS);
  float*  cst  = (float*)(base + OFF_CST);
  __bf16* wf  = (__bf16*)(base + OFF_WF);
  __bf16* cf  = (__bf16*)(base + OFF_CF);
  float*  num = (float*)(base + OFF_NUM);
  float*  den = (float*)(base + OFF_DEN);
  int*    lastTag = (int*)(base + OFF_LT);
  unsigned char* hist = (unsigned char*)(base + OFF_HIST);

  k_prep_w1<<<(KCH * 2048 + 255) / 256, 256, 0, stream>>>(W1, wf);
  k_prep_wc<<<(6 * 512 + 255) / 256, 256, 0, stream>>>(Wc, cf);
  k_gemm1   <<<MTOT / 128, 256, 0, stream>>>(x, wf, b1, h16);
  k_conv_em <<<MTOT / 128, 256, 0, stream>>>(h16, cf, bc, W2, b2, em);
  k_crf_num <<<NB, 256, 0, stream>>>(em, labels, tlen, startv, endv, trans, num);
  k_scan_chunks <<<(NB * CCH + 255) / 256, 256, 0, stream>>>(em, tlen, trans, mats);
  k_scan_combine<<<1, 32, 0, stream>>>(em, startv, endv, mats, cst, den, lastTag);
  k_vit_replay  <<<(NB * CCH + 255) / 256, 256, 0, stream>>>(em, tlen, trans, cst, hist);
  k_backtrace   <<<1, 32, 0, stream>>>(hist, tlen, lastTag, num, den, out);
}
